// GlobalAttention_29257317220740
// MI455X (gfx1250) — matrix operand order repaired and checked by reading, not compiled, not run on hardware
//
#include <hip/hip_runtime.h>

// ---------------------------------------------------------------------------
// GlobalAttention for MI455X (gfx1250, wave32, WMMA).
//
// Math refactor:  res = (q kT) v * scale  ==  feat @ A + c   with
//   S = kT v (256x256), A = scale * WqT @ S, c = scale * (bq @ S)
// => one 65536x256x256 GEMM (bandwidth-bound) instead of two 34-GFLOP GEMMs.
// ---------------------------------------------------------------------------

typedef __attribute__((ext_vector_type(8)))  _Float16 v8h;
typedef __attribute__((ext_vector_type(16))) _Float16 v16h;
typedef __attribute__((ext_vector_type(8)))  float    v8f;

#define NPTS 65536
#define DMODEL 256
#define MSAMP 1024
#define FPS_NB 32
#define FPS_NT 1024

// ------------------------------- init --------------------------------------
__global__ void init_kernel(unsigned* __restrict__ bar, int* __restrict__ idx) {
  int t = threadIdx.x;
  if (t < 1024) bar[t] = 0u;
  if (t == 0) idx[0] = 0;
}

// ------------------------------- FPS ---------------------------------------
__device__ __forceinline__ float bperm_f(int srcLane, float v) {
  return __int_as_float(__builtin_amdgcn_ds_bpermute(srcLane << 2, __float_as_int(v)));
}
__device__ __forceinline__ int bperm_i(int srcLane, int v) {
  return __builtin_amdgcn_ds_bpermute(srcLane << 2, v);
}

__global__ __launch_bounds__(FPS_NT) void fps_kernel(
    const float* __restrict__ xyz, int* __restrict__ idx_out,
    unsigned* __restrict__ bar, float* __restrict__ cand_d, int* __restrict__ cand_i)
{
  const int tid  = threadIdx.x;
  const int bid  = blockIdx.x;
  const int lane = tid & 31;
  const int wv   = tid >> 5;
  const int g    = bid * FPS_NT + tid;       // 0 .. 32767
  const int p0   = g, p1 = g + FPS_NB * FPS_NT;

  __shared__ float sbd[32];
  __shared__ int   sbi[32];
  __shared__ float scx, scy, scz;

  float x0 = xyz[3*p0+0], y0 = xyz[3*p0+1], z0 = xyz[3*p0+2];
  float x1 = xyz[3*p1+0], y1 = xyz[3*p1+1], z1 = xyz[3*p1+2];
  float d0 = 1e10f, d1 = 1e10f;

  if (tid == 0) {
    scx = xyz[0]; scy = xyz[1]; scz = xyz[2];
    if (bid == 0) idx_out[0] = 0;
  }
  __syncthreads();

  for (int it = 0; it < MSAMP - 1; ++it) {
    const float cx = scx, cy = scy, cz = scz;
    // Non-contracted math to bit-match the reference distance computation.
    float ax = __fsub_rn(x0, cx), ay = __fsub_rn(y0, cy), az = __fsub_rn(z0, cz);
    float dd0 = __fadd_rn(__fadd_rn(__fmul_rn(ax,ax), __fmul_rn(ay,ay)), __fmul_rn(az,az));
    float bx = __fsub_rn(x1, cx), by = __fsub_rn(y1, cy), bz = __fsub_rn(z1, cz);
    float dd1 = __fadd_rn(__fadd_rn(__fmul_rn(bx,bx), __fmul_rn(by,by)), __fmul_rn(bz,bz));
    d0 = fminf(d0, dd0);
    d1 = fminf(d1, dd1);

    float bd = d0; int bi = p0;
    if (d1 > d0) { bd = d1; bi = p1; }            // tie -> lower index (p0)

    // wave32 xor-reduce (argmax, tie -> lowest index)
    #pragma unroll
    for (int off = 16; off >= 1; off >>= 1) {
      float od = bperm_f(lane ^ off, bd);
      int   oi = bperm_i(lane ^ off, bi);
      if (od > bd || (od == bd && oi < bi)) { bd = od; bi = oi; }
    }
    if (lane == 0) { sbd[wv] = bd; sbi[wv] = bi; }
    __syncthreads();

    if (tid < 32) {
      bd = sbd[lane]; bi = sbi[lane];
      #pragma unroll
      for (int off = 16; off >= 1; off >>= 1) {
        float od = bperm_f(lane ^ off, bd);
        int   oi = bperm_i(lane ^ off, bi);
        if (od > bd || (od == bd && oi < bi)) { bd = od; bi = oi; }
      }
      if (tid == 0) {
        const int buf = (it & 1) * FPS_NB;
        cand_d[buf + bid] = bd;
        cand_i[buf + bid] = bi;
        __threadfence();
        atomicAdd(&bar[it], 1u);
        while (__atomic_load_n(&bar[it], __ATOMIC_RELAXED) < (unsigned)FPS_NB) {
          __builtin_amdgcn_s_sleep(1);
        }
        __threadfence();
        float wd = -1.0f; int wi = 0x7fffffff;
        #pragma unroll
        for (int b = 0; b < FPS_NB; ++b) {
          float od = cand_d[buf + b]; int oi = cand_i[buf + b];
          if (od > wd || (od == wd && oi < wi)) { wd = od; wi = oi; }
        }
        scx = xyz[3*wi+0]; scy = xyz[3*wi+1]; scz = xyz[3*wi+2];
        if (bid == 0) idx_out[it + 1] = wi;
      }
    }
    __syncthreads();
  }
}

// ------------------- k = f@WkT+bk, v = f@WvT+bv (gather fused) -------------
__global__ __launch_bounds__(256) void kv_kernel(
    const float* __restrict__ feat,
    const float* __restrict__ Wk, const float* __restrict__ bk,
    const float* __restrict__ Wv, const float* __restrict__ bv,
    const int* __restrict__ idx,
    float* __restrict__ kmat, float* __restrict__ vmat)
{
  __shared__ float fls[16][256];
  __shared__ int   sidx[16];
  const int t  = threadIdx.x;
  const int m0 = blockIdx.x * 16;

  if (t < 16) sidx[t] = idx[m0 + t];
  __syncthreads();
  #pragma unroll
  for (int r = 0; r < 16; ++r) fls[r][t] = feat[sidx[r] * DMODEL + t];
  __syncthreads();

  float ak[16], av[16];
  #pragma unroll
  for (int r = 0; r < 16; ++r) { ak[r] = 0.f; av[r] = 0.f; }

  const float* wkr = Wk + t * DMODEL;
  const float* wvr = Wv + t * DMODEL;
  for (int dch = 0; dch < DMODEL; dch += 4) {
    const float4 wk4 = *(const float4*)(wkr + dch);
    const float4 wv4 = *(const float4*)(wvr + dch);
    #pragma unroll
    for (int r = 0; r < 16; ++r) {
      const float4 f4 = *(const float4*)&fls[r][dch];
      ak[r] += wk4.x*f4.x + wk4.y*f4.y + wk4.z*f4.z + wk4.w*f4.w;
      av[r] += wv4.x*f4.x + wv4.y*f4.y + wv4.z*f4.z + wv4.w*f4.w;
    }
  }
  const float bkv = bk[t], bvv = bv[t];
  #pragma unroll
  for (int r = 0; r < 16; ++r) {
    kmat[(m0 + r) * DMODEL + t] = ak[r] + bkv;
    vmat[(m0 + r) * DMODEL + t] = av[r] + bvv;
  }
}

// ------------------------------ S = kT @ v ---------------------------------
__global__ __launch_bounds__(256) void s_kernel(
    const float* __restrict__ kmat, const float* __restrict__ vmat,
    float* __restrict__ S)
{
  __shared__ float kt[32][17], vt[32][17];
  const int t  = threadIdx.x;
  const int i0 = (blockIdx.x >> 4) * 16;
  const int j0 = (blockIdx.x & 15) * 16;
  const int ti = t >> 4, tj = t & 15;

  float acc = 0.f;
  for (int m0 = 0; m0 < MSAMP; m0 += 32) {
    const int e0 = t, e1 = t + 256;
    kt[e0 >> 4][e0 & 15] = kmat[(m0 + (e0 >> 4)) * DMODEL + i0 + (e0 & 15)];
    kt[e1 >> 4][e1 & 15] = kmat[(m0 + (e1 >> 4)) * DMODEL + i0 + (e1 & 15)];
    vt[e0 >> 4][e0 & 15] = vmat[(m0 + (e0 >> 4)) * DMODEL + j0 + (e0 & 15)];
    vt[e1 >> 4][e1 & 15] = vmat[(m0 + (e1 >> 4)) * DMODEL + j0 + (e1 & 15)];
    __syncthreads();
    #pragma unroll
    for (int mm = 0; mm < 32; ++mm) acc += kt[mm][ti] * vt[mm][tj];
    __syncthreads();
  }
  S[(i0 + ti) * DMODEL + (j0 + tj)] = acc;
}

// ----------------- A^T (f16) = (scale * WqT @ S)^T,  c = scale*bq@S --------
__global__ __launch_bounds__(256) void a_kernel(
    const float* __restrict__ Wq, const float* __restrict__ bq,
    const float* __restrict__ S,
    _Float16* __restrict__ At, float* __restrict__ cvec)
{
  const int j = blockIdx.x;      // output column of A  (row of A^T)
  const int d = threadIdx.x;     // K index
  float acc = 0.f, accC = 0.f;
  for (int tt = 0; tt < DMODEL; ++tt) {
    const float s = S[tt * DMODEL + j];
    acc  += Wq[tt * DMODEL + d] * s;
    accC += bq[tt] * s;
  }
  const float scale = 0.0625f;   // 1/sqrt(256)
  At[j * DMODEL + d] = (_Float16)(acc * scale);
  if (d == 0) cvec[j] = accC * scale;
}

// -------- out[:, :256] = feat @ A + c ; out[:, 256:] = feat  (WMMA) --------
__global__ __launch_bounds__(256) void main_gemm(
    const float* __restrict__ feat, const _Float16* __restrict__ At,
    const float* __restrict__ cvec, float* __restrict__ out)
{
  __shared__ _Float16 sF[128 * 264];   // 128 rows, pitch 264 halves (16B mult)
  __shared__ float    sC[256];

  const int tid  = threadIdx.x;
  const int row0 = blockIdx.x * 128;

  sC[tid] = cvec[tid];
  // Stage feat tile as f16 into LDS; fuse the concat copy (out cols 256..511).
  #pragma unroll
  for (int i = 0; i < 32; ++i) {
    const int q  = tid + 256 * i;
    const int r  = q >> 6;
    const int c4 = (q & 63) << 2;
    const float4 f = *(const float4*)(feat + (row0 + r) * DMODEL + c4);
    _Float16* dst = sF + r * 264 + c4;
    dst[0] = (_Float16)f.x; dst[1] = (_Float16)f.y;
    dst[2] = (_Float16)f.z; dst[3] = (_Float16)f.w;
    *(float4*)(out + (row0 + r) * 512 + 256 + c4) = f;
  }
  __syncthreads();

  const int lane = tid & 31, wv = tid >> 5;
  const int wrow = wv & 3;       // 4 row-groups of 32 rows
  const int wcol = wv >> 2;      // 2 col-halves of 128 cols
  const int l15  = lane & 15, lhi = lane >> 4;

  const v8f vzero = {0.f, 0.f, 0.f, 0.f, 0.f, 0.f, 0.f, 0.f};
  v8f acc[2][8];
  #pragma unroll
  for (int m = 0; m < 2; ++m)
    #pragma unroll
    for (int n = 0; n < 8; ++n) acc[m][n] = vzero;

  for (int kc = 0; kc < 8; ++kc) {
    // A-matrix fragments (16x32 f16): lane<16 holds K {0..7,16..23}, lane>=16 {8..15,24..31}
    v16h af[2];
    #pragma unroll
    for (int m = 0; m < 2; ++m) {
      const int rowA = 32 * wrow + 16 * m + l15;
      const _Float16* p = sF + rowA * 264 + kc * 32 + lhi * 8;
      const v8h lo = *(const v8h*)p;
      const v8h hi = *(const v8h*)(p + 16);
      af[m] = __builtin_shufflevector(lo, hi, 0,1,2,3,4,5,6,7,8,9,10,11,12,13,14,15);
    }
    // B-matrix fragments (32x16 f16) from L2-resident A^T: lane<16 holds K {0..7,16..23}, lane>=16 {8..15,24..31}
    v16h bf[8];
    #pragma unroll
    for (int n = 0; n < 8; ++n) {
      const int col = wcol * 128 + n * 16 + l15;
      const _Float16* p = At + col * DMODEL + kc * 32 + lhi * 8;
      __builtin_prefetch(p + 32, 0, 1);   // next K-chunk -> global_prefetch_b8
      const v8h lo = *(const v8h*)p;
      const v8h hi = *(const v8h*)(p + 16);
      bf[n] = __builtin_shufflevector(lo, hi, 0,1,2,3,4,5,6,7,8,9,10,11,12,13,14,15);
    }
    #pragma unroll
    for (int n = 0; n < 8; ++n) {
      acc[0][n] = __builtin_amdgcn_wmma_f32_16x16x32_f16(
          false, af[0], false, bf[n], (short)0, acc[0][n], false, false);
      acc[1][n] = __builtin_amdgcn_wmma_f32_16x16x32_f16(
          false, af[1], false, bf[n], (short)0, acc[1][n], false, false);
    }
  }

  // Epilogue: C/D layout -> lane<16: rows m0..m0+7 (VGPR r), lane>=16: rows m0+8..15
  #pragma unroll
  for (int m = 0; m < 2; ++m) {
    #pragma unroll
    for (int n = 0; n < 8; ++n) {
      const int col   = wcol * 128 + n * 16 + l15;
      const float cad = sC[col];
      const int rbase = row0 + 32 * wrow + 16 * m + lhi * 8;
      #pragma unroll
      for (int r = 0; r < 8; ++r) {
        out[(rbase + r) * 512 + col] = acc[m][n][r] + cad;
      }
    }
  }
}

// ---------------------------------------------------------------------------
extern "C" void kernel_launch(void* const* d_in, const int* in_sizes, int n_in,
                              void* d_out, int out_size, void* d_ws, size_t ws_size,
                              hipStream_t stream)
{
  (void)in_sizes; (void)n_in; (void)out_size; (void)ws_size;
  const float* p_xyz = (const float*)d_in[0];
  const float* feat  = (const float*)d_in[1];
  const float* Wq    = (const float*)d_in[2];
  const float* bq    = (const float*)d_in[3];
  const float* Wk    = (const float*)d_in[4];
  const float* bk    = (const float*)d_in[5];
  const float* Wv    = (const float*)d_in[6];
  const float* bv    = (const float*)d_in[7];
  float* out = (float*)d_out;

  char* ws = (char*)d_ws;
  int*      idx    = (int*)(ws + 0);                       // 4 KB
  unsigned* bar    = (unsigned*)(ws + 4096);               // 4 KB
  float*    cand_d = (float*)(ws + 8192);                  // 256 B (2x32)
  int*      cand_i = (int*)(ws + 8192 + 256);              // 256 B
  float*    kmat   = (float*)(ws + 16384);                 // 1 MB
  float*    vmat   = (float*)(ws + 16384 + 1048576);       // 1 MB
  float*    Smat   = (float*)(ws + 16384 + 2097152);       // 256 KB
  _Float16* At     = (_Float16*)(ws + 16384 + 2097152 + 262144);           // 128 KB
  float*    cvec   = (float*)(ws + 16384 + 2097152 + 262144 + 131072);     // 1 KB

  init_kernel<<<1, 1024, 0, stream>>>(bar, idx);
  fps_kernel<<<FPS_NB, FPS_NT, 0, stream>>>(p_xyz, idx, bar, cand_d, cand_i);
  kv_kernel<<<MSAMP / 16, 256, 0, stream>>>(feat, Wk, bk, Wv, bv, idx, kmat, vmat);
  s_kernel<<<256, 256, 0, stream>>>(kmat, vmat, Smat);
  a_kernel<<<256, 256, 0, stream>>>(Wq, bq, Smat, At, cvec);
  main_gemm<<<NPTS / 128, 256, 0, stream>>>(feat, At, cvec, out);
}
